// VisionAttention_15642270892547
// MI455X (gfx1250) — compile-verified, matrix operand order checked by reading, not run on hardware
//
#include <hip/hip_runtime.h>

typedef __attribute__((ext_vector_type(16))) __bf16 v16bf;
typedef __attribute__((ext_vector_type(8)))  __bf16 v8bf;
typedef __attribute__((ext_vector_type(8)))  float  v8f;

#define SS 4096
#define HH 16
#define DD 64
#define NEGF (-1.0e30f)

__device__ __forceinline__ int seg_of(int i, const int* cu) {
  int s = 0;
#pragma unroll
  for (int j = 1; j <= 8; ++j) s += (i >= cu[j]) ? 1 : 0;
  return s;
}

__device__ __forceinline__ v16bf frag_cat(v8bf a, v8bf b) {
  return __builtin_shufflevector(a, b, 0,1,2,3,4,5,6,7,8,9,10,11,12,13,14,15);
}

// 16-value fragment at p[0..7], p[16..23]: bf16 path = two 16B loads, no cvt.
__device__ __forceinline__ v16bf load16f(const __bf16* __restrict__ p, float) {
  return frag_cat(*(const v8bf*)p, *(const v8bf*)(p + 16));
}
__device__ __forceinline__ v16bf load16f(const float* __restrict__ p, float s) {
  v16bf f;
#pragma unroll
  for (int i = 0; i < 8; ++i) {
    f[i]     = (__bf16)(p[i] * s);
    f[i + 8] = (__bf16)(p[i + 16] * s);
  }
  return f;
}

// Per-lane V slice for one 32x64 tile: tokens tq,tq+4,..,tq+28 ; d in [d0,d0+8).
__device__ __forceinline__ void loadVregs(const __bf16* __restrict__ p, v8bf vr[8]) {
#pragma unroll
  for (int i = 0; i < 8; ++i) vr[i] = *(const v8bf*)(p + (size_t)i * 4 * HH * DD);
}
__device__ __forceinline__ void loadVregs(const float* __restrict__ p, v8bf vr[8]) {
#pragma unroll
  for (int i = 0; i < 8; ++i) {
    const float* q = p + (size_t)i * 4 * HH * DD;
    v8bf x;
#pragma unroll
    for (int j = 0; j < 8; ++j) x[j] = (__bf16)q[j];
    vr[i] = x;
  }
}

// Pre-pass: one-shot f32 -> bf16 (Q pre-scaled by 1/sqrt(D)).
__global__ void cvt_bf16(const float* __restrict__ q, const float* __restrict__ k,
                         const float* __restrict__ v, __bf16* __restrict__ qb,
                         __bf16* __restrict__ kb, __bf16* __restrict__ vb, int n4) {
  const int stride = gridDim.x * blockDim.x;
  for (int i = blockIdx.x * blockDim.x + threadIdx.x; i < n4; i += stride) {
    const float4 a = ((const float4*)q)[i];
    const float4 b = ((const float4*)k)[i];
    const float4 c = ((const float4*)v)[i];
    qb[4*i+0] = (__bf16)(a.x*0.125f); qb[4*i+1] = (__bf16)(a.y*0.125f);
    qb[4*i+2] = (__bf16)(a.z*0.125f); qb[4*i+3] = (__bf16)(a.w*0.125f);
    kb[4*i+0] = (__bf16)b.x; kb[4*i+1] = (__bf16)b.y;
    kb[4*i+2] = (__bf16)b.z; kb[4*i+3] = (__bf16)b.w;
    vb[4*i+0] = (__bf16)c.x; vb[4*i+1] = (__bf16)c.y;
    vb[4*i+2] = (__bf16)c.z; vb[4*i+3] = (__bf16)c.w;
  }
}

template <typename T>
__global__ __launch_bounds__(32)
void vision_attn_wmma(const T* __restrict__ q, const T* __restrict__ k,
                      const T* __restrict__ v, const int* __restrict__ cu9,
                      float* __restrict__ out, float qscale) {
  __shared__ __align__(16) __bf16 VT[64][32];      // V^T tile (B frags, P*V)
  __shared__ __align__(16) __bf16 Pt[2][16][32];   // P relayout C->A per sub-tile

  const int lane = threadIdx.x;
  const int l16  = lane & 15;
  const int half = lane >> 4;
  const int m0   = blockIdx.x * 32;
  const int h    = blockIdx.y;

  // cu_seqlens is wave-uniform: pin to SGPRs so all derived bounds are scalar.
  int cu[9];
#pragma unroll
  for (int i = 0; i < 9; ++i) cu[i] = __builtin_amdgcn_readfirstlane(cu9[i]);

  int uni[2], uLo[2], uHi[2];
#pragma unroll
  for (int mt = 0; mt < 2; ++mt) {
    const int sA = seg_of(m0 + mt * 16, cu), sB = seg_of(m0 + mt * 16 + 15, cu);
    uni[mt] = (sA == sB);
    uLo[mt] = cu[sA];
    uHi[mt] = cu[sB + 1];
  }
  const int kvHi   = cu[seg_of(m0 + 31, cu) + 1];
  const int nStart = cu[seg_of(m0, cu)] & ~31;

  v16bf aq[2][2];
#pragma unroll
  for (int mt = 0; mt < 2; ++mt) {
    const T* qrow = q + ((size_t)(m0 + mt * 16 + l16) * HH + h) * DD;
#pragma unroll
    for (int db = 0; db < 2; ++db)
      aq[mt][db] = load16f(qrow + db * 32 + half * 8, qscale);
  }

  v16bf ones;   // constant B operand: l += P*1 via WMMA
#pragma unroll
  for (int i = 0; i < 16; ++i) ones[i] = (__bf16)1.0f;

  v8f acc[2][4], lacc[2];
  float mh[2] = {NEGF, NEGF};
#pragma unroll
  for (int mt = 0; mt < 2; ++mt) {
#pragma unroll
    for (int r = 0; r < 8; ++r) lacc[mt][r] = 0.0f;
#pragma unroll
    for (int dc = 0; dc < 4; ++dc)
#pragma unroll
      for (int r = 0; r < 8; ++r) acc[mt][dc][r] = 0.0f;
  }

  // ---- software-pipelined streams: K fragments + V slice, one window ahead ----
  const int tq = lane >> 3, d0 = (lane & 7) * 8;
  const size_t kstep = (size_t)32 * HH * DD;
  const T* krow0 = k + ((size_t)(nStart + l16) * HH + h) * DD;
  const T* krow1 = krow0 + (size_t)16 * HH * DD;
  const T* vptr  = v + ((size_t)(nStart + tq) * HH + h) * DD + d0;

  v16bf kb0[2], kb1[2];
  v8bf  vr[8];
#pragma unroll
  for (int db = 0; db < 2; ++db) {
    kb0[db] = load16f(krow0 + db * 32 + half * 8, 1.0f);
    kb1[db] = load16f(krow1 + db * 32 + half * 8, 1.0f);
  }
  loadVregs(vptr, vr);
  krow0 += kstep; krow1 += kstep; vptr += kstep;

  for (int n0 = nStart; n0 < kvHi; n0 += 32) {
    // ---- commit prefetched V slice to LDS (transposed) ----
#pragma unroll
    for (int i = 0; i < 8; ++i)
#pragma unroll
      for (int j = 0; j < 8; ++j) VT[d0 + j][i * 4 + tq] = vr[i][j];
    __syncthreads();

    // ---- scores from prefetched K fragments (no memory wait here) ----
    v8f c[2][2];
#pragma unroll
    for (int mt = 0; mt < 2; ++mt)
#pragma unroll
      for (int nt = 0; nt < 2; ++nt)
#pragma unroll
        for (int r = 0; r < 8; ++r) c[mt][nt][r] = 0.0f;
#pragma unroll
    for (int db = 0; db < 2; ++db)
#pragma unroll
      for (int mt = 0; mt < 2; ++mt) {
        c[mt][0] = __builtin_amdgcn_wmma_f32_16x16x32_bf16(false, aq[mt][db], false, kb0[db],
                                                           (short)0, c[mt][0], false, false);
        c[mt][1] = __builtin_amdgcn_wmma_f32_16x16x32_bf16(false, aq[mt][db], false, kb1[db],
                                                           (short)0, c[mt][1], false, false);
      }

    // ---- issue next window's K/V loads; they complete under softmax+PV ----
    if (n0 + 32 < kvHi) {
#pragma unroll
      for (int db = 0; db < 2; ++db) {
        kb0[db] = load16f(krow0 + db * 32 + half * 8, 1.0f);
        kb1[db] = load16f(krow1 + db * 32 + half * 8, 1.0f);
      }
      loadVregs(vptr, vr);
      krow0 += kstep; krow1 += kstep; vptr += kstep;
    }

    // ---- mask (uniform fast path; cold path recomputes segment ids) ----
    const int col0 = n0 + l16;
    const int col1 = col0 + 16;
    float al[2];
#pragma unroll
    for (int mt = 0; mt < 2; ++mt) {
      float x0[8], x1[8];
      if (uni[mt] && n0 >= uLo[mt] && n0 + 32 <= uHi[mt]) {   // uniform fast path
#pragma unroll
        for (int r = 0; r < 8; ++r) { x0[r] = c[mt][0][r]; x1[r] = c[mt][1][r]; }
      } else {                                                // rare boundary path
        const int sc0 = seg_of(col0, cu);
        const int sc1 = seg_of(col1, cu);
#pragma unroll
        for (int r = 0; r < 8; ++r) {
          const int sr = seg_of(m0 + mt * 16 + half * 8 + r, cu);
          x0[r] = (sr == sc0) ? c[mt][0][r] : NEGF;
          x1[r] = (sr == sc1) ? c[mt][1][r] : NEGF;
        }
      }
      float mloc = NEGF;
#pragma unroll
      for (int r = 0; r < 8; ++r) mloc = fmaxf(mloc, fmaxf(x0[r], x1[r]));
#pragma unroll
      for (int off = 1; off < 16; off <<= 1) mloc = fmaxf(mloc, __shfl_xor(mloc, off));

      const float mnew = fmaxf(mh[mt], mloc);
      const float safe = (mnew <= 0.5f * NEGF) ? 0.0f : mnew;
      al[mt] = __expf(mh[mt] - safe);
      mh[mt] = mnew;
#pragma unroll
      for (int r = 0; r < 8; ++r) {
        Pt[mt][half * 8 + r][l16]      = (__bf16)__expf(x0[r] - safe);
        Pt[mt][half * 8 + r][16 + l16] = (__bf16)__expf(x1[r] - safe);
      }
    }

    if (__any((al[0] != 1.0f) || (al[1] != 1.0f))) {   // rescale only on new max
#pragma unroll
      for (int mt = 0; mt < 2; ++mt) {
#pragma unroll
        for (int r = 0; r < 8; ++r) lacc[mt][r] *= al[mt];
#pragma unroll
        for (int dc = 0; dc < 4; ++dc)
#pragma unroll
          for (int r = 0; r < 8; ++r) acc[mt][dc][r] *= al[mt];
      }
    }
    __syncthreads();

    // ---- O += P*V ; l += P*1 ; V B-frags shared by both sub-tiles ----
    const v16bf ap0 = frag_cat(*(const v8bf*)&Pt[0][l16][half * 8],
                               *(const v8bf*)&Pt[0][l16][half * 8 + 16]);
    const v16bf ap1 = frag_cat(*(const v8bf*)&Pt[1][l16][half * 8],
                               *(const v8bf*)&Pt[1][l16][half * 8 + 16]);
    lacc[0] = __builtin_amdgcn_wmma_f32_16x16x32_bf16(false, ap0, false, ones,
                                                      (short)0, lacc[0], false, false);
    lacc[1] = __builtin_amdgcn_wmma_f32_16x16x32_bf16(false, ap1, false, ones,
                                                      (short)0, lacc[1], false, false);
#pragma unroll
    for (int dc = 0; dc < 4; ++dc) {
      const v16bf bv = frag_cat(*(const v8bf*)&VT[dc * 16 + l16][half * 8],
                                *(const v8bf*)&VT[dc * 16 + l16][half * 8 + 16]);
      acc[0][dc] = __builtin_amdgcn_wmma_f32_16x16x32_bf16(false, ap0, false, bv,
                                                           (short)0, acc[0][dc], false, false);
      acc[1][dc] = __builtin_amdgcn_wmma_f32_16x16x32_bf16(false, ap1, false, bv,
                                                           (short)0, acc[1][dc], false, false);
    }
    __syncthreads();
  }

#pragma unroll
  for (int mt = 0; mt < 2; ++mt)
#pragma unroll
    for (int r = 0; r < 8; ++r) {
      const int row = m0 + mt * 16 + half * 8 + r;
      const float inv = (lacc[mt][r] > 0.0f) ? (1.0f / lacc[mt][r]) : 0.0f;
      float* orow = out + ((size_t)row * HH + h) * DD;
#pragma unroll
      for (int dc = 0; dc < 4; ++dc) orow[dc * 16 + l16] = acc[mt][dc][r] * inv;
    }
}

extern "C" void kernel_launch(void* const* d_in, const int* in_sizes, int n_in,
                              void* d_out, int out_size, void* d_ws, size_t ws_size,
                              hipStream_t stream) {
  (void)in_sizes; (void)n_in; (void)out_size;
  const float* q  = (const float*)d_in[0];
  const float* k  = (const float*)d_in[1];
  const float* v  = (const float*)d_in[2];
  const int*   cu = (const int*)d_in[3];
  float* out = (float*)d_out;

  const size_t nElem = (size_t)SS * HH * DD;        // 4.19M per tensor
  dim3 grid(SS / 32, HH);
  dim3 block(32);

  if (ws_size >= 3 * nElem * sizeof(__bf16)) {
    __bf16* qb = (__bf16*)d_ws;
    __bf16* kb = qb + nElem;
    __bf16* vb = kb + nElem;
    cvt_bf16<<<1024, 256, 0, stream>>>(q, k, v, qb, kb, vb, (int)(nElem / 4));
    vision_attn_wmma<__bf16><<<grid, block, 0, stream>>>(qb, kb, vb, cu, out, 1.0f);
  } else {
    vision_attn_wmma<float><<<grid, block, 0, stream>>>(q, k, v, cu, out, 0.125f);
  }
}
